// FixedAttention_77756087927151
// MI455X (gfx1250) — compile-verified, matrix operand order checked by reading, not run on hardware
//
#include <hip/hip_runtime.h>
#include <hip/hip_bf16.h>

// ---------------------------------------------------------------------------
// Fixed attention block for MI455X (gfx1250), wave32 + WMMA f32_16x16x32_f16,
// with GLOBAL_LOAD_ASYNC_TO_LDS_B128 tile staging (ASYNCcnt).
//   B=2, S=2048, D=1024, H=16, HD=64
// Pipeline:
//   1) cvt: fp32 -> f16 for x, Wq(*0.125), Wk, Wv, Wo
//   2) gemm: Q=x Wq^T, K=x Wk^T, V=x Wv^T  (f16 out, [B,H,S,HD] layout)
//   3) flash attention per (b,h,qtile) -> Oh [B,S,D] f16
//   4) gemm: out = Oh Wo^T + b_o (fp32 out)
// ---------------------------------------------------------------------------

typedef _Float16 half_t;
typedef __attribute__((ext_vector_type(16))) _Float16 v16h_t;
typedef __attribute__((ext_vector_type(8)))  _Float16 v8h_t;
typedef __attribute__((ext_vector_type(4)))  _Float16 v4h_t;
typedef __attribute__((ext_vector_type(8)))  float    v8f_t;

#define BATCH 2
#define SEQ   2048
#define DIM   1024
#define NHEAD 16
#define HDIM  64

// ---------------------------------------------------------------------------
// WMMA wrapper: D = A(16x32 f16) * B(32x16 f16) + C(16x16 f32)
// ---------------------------------------------------------------------------
__device__ inline v8f_t wmma_f32_f16(v16h_t a, v16h_t b, v8f_t c) {
  return __builtin_amdgcn_wmma_f32_16x16x32_f16(
      /*neg_a=*/false, a, /*neg_b=*/false, b,
      /*c_mod=*/(short)0, c, /*reuse_a=*/false, /*reuse_b=*/false);
}

// ---------------------------------------------------------------------------
// Async global -> LDS copy, 16 bytes per lane (GLOBAL_LOAD_ASYNC_TO_LDS_B128,
// tracked by ASYNCcnt; data lands in LDS without touching VGPRs).
// LDS operand: byte offset within the workgroup's LDS allocation == low 32
// bits of the generic shared-memory pointer (shared aperture is in the high
// 32 bits per the ISA aperture check).
// ---------------------------------------------------------------------------
__device__ inline void async_copy_b128(void* lds_dst, const void* gsrc) {
  unsigned lds_off = (unsigned)(unsigned long long)lds_dst;
  asm volatile("global_load_async_to_lds_b128 %0, %1, off"
               :: "v"(lds_off), "v"(gsrc) : "memory");
}
__device__ inline void wait_async_all() {
  asm volatile("s_wait_asynccnt 0x0" ::: "memory");
}

// ---------------------------------------------------------------------------
// Build a 16x32 f16 fragment (A-layout per ISA 7.12.2) from a row-major tile
// in LDS.  Lane L (L<16) holds row M=L, K = {0..7, 16..23}; lane L+16 holds
// row M=L, K = {8..15, 24..31}.  Two 16B LDS loads per fragment.
// B-fragments (32x16, K x N) use the same call on an N-major (transposed)
// tile: lane = N column, 16 K values per lane with the same split.
// ---------------------------------------------------------------------------
__device__ inline v16h_t load_frag(const half_t* base, int row_stride) {
  int lane = threadIdx.x & 31;
  int m    = lane & 15;
  int hi   = lane >> 4;
  const half_t* p = base + (size_t)m * row_stride + hi * 8;
  v8h_t lo = *(const v8h_t*)(p);        // K = hi*8 .. hi*8+7
  v8h_t hp = *(const v8h_t*)(p + 16);   // K = 16+hi*8 .. 16+hi*8+7
  v16h_t r;
#pragma unroll
  for (int i = 0; i < 8; ++i) { r[i] = lo[i]; r[i + 8] = hp[i]; }
  return r;
}

// ---------------------------------------------------------------------------
// fp32 -> f16 conversion (with optional scale), 4 elements / thread
// ---------------------------------------------------------------------------
__global__ __launch_bounds__(256) void cvt_f32_f16_kernel(const float* __restrict__ in,
                                                          half_t* __restrict__ out,
                                                          float scale, int n4) {
  int i = blockIdx.x * blockDim.x + threadIdx.x;
  if (i < n4) {
    float4 f = ((const float4*)in)[i];
    v4h_t h;
    h[0] = (half_t)(f.x * scale);
    h[1] = (half_t)(f.y * scale);
    h[2] = (half_t)(f.z * scale);
    h[3] = (half_t)(f.w * scale);
    ((v4h_t*)out)[i] = h;
  }
}

// ---------------------------------------------------------------------------
// GEMM: C[r][n] = sum_k X[r][k] * W[n][k],  R=4096, K=1024, N=1024.
// Block tile 128x64, BK=64.  8 waves in a 4(M) x 2(N) grid; each wave owns a
// 32x32 tile = 2x2x2 v_wmma_f32_16x16x32_f16 per K step (8 WMMA / barrier).
// Tiles staged via GLOBAL_LOAD_ASYNC_TO_LDS_B128.
// mode 0: f16 output scattered to head layout [B,H,S,HD]
// mode 1: fp32 output + bias, flat [R,N]
// ---------------------------------------------------------------------------
__global__ __launch_bounds__(256) void gemm_xwT_kernel(const half_t* __restrict__ X,
                                                       const half_t* __restrict__ W,
                                                       void* __restrict__ out,
                                                       const float* __restrict__ bias,
                                                       int mode) {
  const int K = DIM;
  int bm = blockIdx.x;          // 0..31  (row tiles of 128)
  int bn = blockIdx.y;          // 0..15  (col tiles of 64)
  int tid = threadIdx.x;
  int wid = tid >> 5;
  int lane = tid & 31;
  int wm = wid & 3;             // wave row   (4)
  int wn = wid >> 2;            // wave col   (2)
  int m16 = lane & 15;
  int hi  = lane >> 4;

  __shared__ half_t As[128 * 64];   // 16 KB
  __shared__ half_t Bs[64 * 64];    //  8 KB

  v8f_t acc[2][2] = {};

  const half_t* Xb = X + (size_t)(bm * 128) * K;
  const half_t* Wb = W + (size_t)(bn * 64) * K;

  for (int k0 = 0; k0 < K; k0 += 64) {
    __syncthreads();    // previous tile fully consumed before overwrite
    // A tile: 128 rows x 64 halves = 1024 chunks of 8 halves (16B) -> 4/thread
#pragma unroll
    for (int c = tid; c < 1024; c += 256) {
      int row = c >> 3;
      int col = (c & 7) * 8;
      async_copy_b128(&As[row * 64 + col], &Xb[(size_t)row * K + k0 + col]);
    }
    // B tile: 64 rows x 64 halves = 512 chunks -> 2/thread
#pragma unroll
    for (int c = tid; c < 512; c += 256) {
      int row = c >> 3;
      int col = (c & 7) * 8;
      async_copy_b128(&Bs[row * 64 + col], &Wb[(size_t)row * K + k0 + col]);
    }
    if (k0 + 64 < K) {
      // warm next tiles in the cache hierarchy (global_prefetch_b8)
      __builtin_prefetch(&Xb[(size_t)(tid >> 1) * K + k0 + 64], 0, 1);
      __builtin_prefetch(&Wb[(size_t)(tid >> 2) * K + k0 + 64], 0, 1);
    }
    wait_async_all();
    __syncthreads();

#pragma unroll
    for (int kc = 0; kc < 2; ++kc) {
      v16h_t a0 = load_frag(&As[(wm * 32 + 0)  * 64 + kc * 32], 64);
      v16h_t a1 = load_frag(&As[(wm * 32 + 16) * 64 + kc * 32], 64);
      v16h_t b0 = load_frag(&Bs[(wn * 32 + 0)  * 64 + kc * 32], 64);
      v16h_t b1 = load_frag(&Bs[(wn * 32 + 16) * 64 + kc * 32], 64);
      acc[0][0] = wmma_f32_f16(a0, b0, acc[0][0]);
      acc[0][1] = wmma_f32_f16(a0, b1, acc[0][1]);
      acc[1][0] = wmma_f32_f16(a1, b0, acc[1][0]);
      acc[1][1] = wmma_f32_f16(a1, b1, acc[1][1]);
    }
  }

#pragma unroll
  for (int tm = 0; tm < 2; ++tm)
#pragma unroll
    for (int tn = 0; tn < 2; ++tn)
#pragma unroll
      for (int i = 0; i < 8; ++i) {
        int r = bm * 128 + wm * 32 + tm * 16 + i + 8 * hi;
        int n = bn * 64  + wn * 32 + tn * 16 + m16;
        float v = acc[tm][tn][i];
        if (mode == 0) {
          int b = r >> 11;            // r / SEQ
          int s = r & (SEQ - 1);
          int hh = n >> 6;            // n / HDIM
          int d  = n & (HDIM - 1);
          ((half_t*)out)[(((size_t)(b * NHEAD + hh) * SEQ) + s) * HDIM + d] = (half_t)v;
        } else {
          ((float*)out)[(size_t)r * DIM + n] = v + bias[n];
        }
      }
}

// ---------------------------------------------------------------------------
// Flash attention.  Grid: (B*H, S/64).  Block: 128 threads = 4 waves; each
// wave owns a 16-row q strip.  Per 64-key tile: scores via WMMA, mask +
// online softmax, P through LDS (C-layout -> A-layout), P*V via WMMA.
// Q/K tiles staged with async-to-LDS; V is transposed manually into Vt.
// Softmax scale is pre-folded into Q (Wq * 0.125 at conversion time).
// ---------------------------------------------------------------------------
__global__ __launch_bounds__(128) void attn_kernel(const half_t* __restrict__ Qh,
                                                   const half_t* __restrict__ Kh,
                                                   const half_t* __restrict__ Vh,
                                                   const int* __restrict__ Mmask,
                                                   half_t* __restrict__ Oh) {
  int bh = blockIdx.x;                 // 0..31
  int b = bh >> 4;
  int h = bh & (NHEAD - 1);
  int qtile = blockIdx.y;              // 0..31
  int tid = threadIdx.x;
  int wid = tid >> 5;
  int lane = tid & 31;
  int m16 = lane & 15;
  int hi  = lane >> 4;

  __shared__ half_t Qs[64 * 64];       // [q][d]
  __shared__ half_t Ks[64 * 64];       // [key][d]
  __shared__ half_t Vt[64 * 64];       // [d][key]  (transposed V)
  __shared__ half_t Ps[4][16 * 64];    // per-wave P staging [qrow][key]

  const half_t* Qbase = Qh + ((size_t)bh * SEQ + (size_t)qtile * 64) * HDIM;
  const half_t* Kbase = Kh + (size_t)bh * SEQ * HDIM;
  const half_t* Vbase = Vh + (size_t)bh * SEQ * HDIM;

  // Q tile 64x64 via async-to-LDS
#pragma unroll
  for (int c = tid; c < 512; c += 128)
    async_copy_b128(&Qs[c * 8], &Qbase[c * 8]);
  wait_async_all();
  __syncthreads();

  v16h_t aq0 = load_frag(&Qs[wid * 16 * 64 + 0],  64);  // d = 0..31
  v16h_t aq1 = load_frag(&Qs[wid * 16 * 64 + 32], 64);  // d = 32..63

  v8f_t Oacc[4] = {};
  float mrow[8], lrow[8];
#pragma unroll
  for (int i = 0; i < 8; ++i) { mrow[i] = -3.0e38f; lrow[i] = 0.0f; }

  // mask row base for this lane group: rows q = qtile*64 + wid*16 + 8*hi + i
  const int* Mrow = Mmask + ((size_t)b * SEQ + (size_t)(qtile * 64 + wid * 16 + 8 * hi)) * SEQ;

  for (int kt = 0; kt < SEQ / 64; ++kt) {
    __syncthreads();
    // K tile 64x64 (row-major [key][d]) via async-to-LDS
#pragma unroll
    for (int c = tid; c < 512; c += 128)
      async_copy_b128(&Ks[c * 8], &Kbase[(size_t)kt * 64 * HDIM + c * 8]);
    // V tile, transposed into Vt[d][key] (manual: async DMA cannot transpose)
#pragma unroll
    for (int c = tid; c < 512; c += 128) {
      int key = c >> 3;
      int d0  = (c & 7) * 8;
      v8h_t v = *(const v8h_t*)&Vbase[(size_t)kt * 64 * HDIM + (size_t)key * HDIM + d0];
#pragma unroll
      for (int e = 0; e < 8; ++e) Vt[(d0 + e) * 64 + key] = v[e];
    }
    wait_async_all();
    __syncthreads();

    // --- scores: S[16q x 64k] ---
    v8f_t Sacc[4] = {};
#pragma unroll
    for (int jn = 0; jn < 4; ++jn) {
      v16h_t bk0 = load_frag(&Ks[jn * 16 * 64 + 0],  64);
      v16h_t bk1 = load_frag(&Ks[jn * 16 * 64 + 32], 64);
      Sacc[jn] = wmma_f32_f16(aq0, bk0, Sacc[jn]);
      Sacc[jn] = wmma_f32_f16(aq1, bk1, Sacc[jn]);
    }

    // --- mask + online softmax (rows live across 16-lane groups) ---
    float corr[8];
#pragma unroll
    for (int i = 0; i < 8; ++i) {
      float rmax = -3.0e38f;
#pragma unroll
      for (int jn = 0; jn < 4; ++jn) {
        int k = kt * 64 + jn * 16 + m16;
        int mv = Mrow[(size_t)i * SEQ + k];
        float s = Sacc[jn][i];
        s = mv ? s : -1.0e9f;
        Sacc[jn][i] = s;
        rmax = fmaxf(rmax, s);
      }
#pragma unroll
      for (int off = 8; off >= 1; off >>= 1)
        rmax = fmaxf(rmax, __shfl_xor(rmax, off, 32));
      float mnew = fmaxf(mrow[i], rmax);
      float c2 = __expf(mrow[i] - mnew);
      mrow[i] = mnew;
      corr[i] = c2;
      float rsum = 0.0f;
#pragma unroll
      for (int jn = 0; jn < 4; ++jn) {
        float p = __expf(Sacc[jn][i] - mnew);
        Sacc[jn][i] = p;
        rsum += p;
      }
#pragma unroll
      for (int off = 8; off >= 1; off >>= 1)
        rsum += __shfl_xor(rsum, off, 32);
      lrow[i] = lrow[i] * c2 + rsum;
    }

    // --- P: C-layout -> A-layout via LDS (wave-private buffer) ---
#pragma unroll
    for (int jn = 0; jn < 4; ++jn)
#pragma unroll
      for (int i = 0; i < 8; ++i)
        Ps[wid][(i + 8 * hi) * 64 + jn * 16 + m16] = (half_t)Sacc[jn][i];
    asm volatile("s_wait_dscnt 0x0" ::: "memory");   // same-wave LDS RAW fence

    // rescale running output
#pragma unroll
    for (int jn = 0; jn < 4; ++jn)
#pragma unroll
      for (int i = 0; i < 8; ++i)
        Oacc[jn][i] *= corr[i];

    v16h_t ap0 = load_frag(&Ps[wid][0],  64);  // keys 0..31
    v16h_t ap1 = load_frag(&Ps[wid][32], 64);  // keys 32..63
#pragma unroll
    for (int jn = 0; jn < 4; ++jn) {
      v16h_t bv0 = load_frag(&Vt[jn * 16 * 64 + 0],  64);
      v16h_t bv1 = load_frag(&Vt[jn * 16 * 64 + 32], 64);
      Oacc[jn] = wmma_f32_f16(ap0, bv0, Oacc[jn]);
      Oacc[jn] = wmma_f32_f16(ap1, bv1, Oacc[jn]);
    }
  }

  // --- write Oh[b, q, h*64 + d] ---
#pragma unroll
  for (int jn = 0; jn < 4; ++jn) {
    int d = jn * 16 + m16;
#pragma unroll
    for (int i = 0; i < 8; ++i) {
      int q = qtile * 64 + wid * 16 + i + 8 * hi;
      float inv = lrow[i] > 0.0f ? 1.0f / lrow[i] : 0.0f;
      Oh[((size_t)b * SEQ + q) * DIM + h * HDIM + d] = (half_t)(Oacc[jn][i] * inv);
    }
  }
}

// ---------------------------------------------------------------------------
// Launch
// ---------------------------------------------------------------------------
extern "C" void kernel_launch(void* const* d_in, const int* in_sizes, int n_in,
                              void* d_out, int out_size, void* d_ws, size_t ws_size,
                              hipStream_t stream) {
  const float* x   = (const float*)d_in[0];
  const int*   M   = (const int*)d_in[1];
  const float* Wq  = (const float*)d_in[2];
  const float* Wk  = (const float*)d_in[3];
  const float* Wv  = (const float*)d_in[4];
  const float* Wo  = (const float*)d_in[5];
  const float* bo  = (const float*)d_in[6];

  char* ws = (char*)d_ws;
  const size_t MB = 1u << 20;
  half_t* xh  = (half_t*)(ws + 0 * MB);   // 8 MB  [B*S, D]
  half_t* wqh = (half_t*)(ws + 8 * MB);   // 2 MB
  half_t* wkh = (half_t*)(ws + 10 * MB);  // 2 MB
  half_t* wvh = (half_t*)(ws + 12 * MB);  // 2 MB
  half_t* woh = (half_t*)(ws + 14 * MB);  // 2 MB
  half_t* Qh  = (half_t*)(ws + 16 * MB);  // 8 MB  [B,H,S,HD]
  half_t* Kh  = (half_t*)(ws + 24 * MB);  // 8 MB
  half_t* Vh  = (half_t*)(ws + 32 * MB);  // 8 MB
  half_t* Oh  = (half_t*)(ws + 40 * MB);  // 8 MB  [B,S,D]

  const int nX4 = BATCH * SEQ * DIM / 4;  // 1048576
  const int nW4 = DIM * DIM / 4;          // 262144
  cvt_f32_f16_kernel<<<nX4 / 256, 256, 0, stream>>>(x,  xh,  1.0f,   nX4);
  cvt_f32_f16_kernel<<<nW4 / 256, 256, 0, stream>>>(Wq, wqh, 0.125f, nW4);  // fold HD^-0.5
  cvt_f32_f16_kernel<<<nW4 / 256, 256, 0, stream>>>(Wk, wkh, 1.0f,   nW4);
  cvt_f32_f16_kernel<<<nW4 / 256, 256, 0, stream>>>(Wv, wvh, 1.0f,   nW4);
  cvt_f32_f16_kernel<<<nW4 / 256, 256, 0, stream>>>(Wo, woh, 1.0f,   nW4);

  dim3 gg(BATCH * SEQ / 128, DIM / 64);   // (32, 16)
  gemm_xwT_kernel<<<gg, 256, 0, stream>>>(xh, wqh, Qh, nullptr, 0);
  gemm_xwT_kernel<<<gg, 256, 0, stream>>>(xh, wkh, Kh, nullptr, 0);
  gemm_xwT_kernel<<<gg, 256, 0, stream>>>(xh, wvh, Vh, nullptr, 0);

  attn_kernel<<<dim3(BATCH * NHEAD, SEQ / 64), 128, 0, stream>>>(Qh, Kh, Vh, M, Oh);

  gemm_xwT_kernel<<<gg, 256, 0, stream>>>(Oh, woh, d_out, bo, 1);
}
